// MHSA_11115375362385
// MI455X (gfx1250) — compile-verified, matrix operand order checked by reading, not run on hardware
//
#include <hip/hip_runtime.h>
#include <hip/hip_bf16.h>

// ---------------------------------------------------------------------------
// MHSA for MI455X (gfx1250, wave32, WMMA). bf16 matrix path, fp32 accumulate.
//   D_MODEL=1024, N_HEADS=16, D_HEAD=64, B=4, S=2048
// GEMMs: 256x64 block tile, 8 waves; W slab staged in LDS via
// GLOBAL_LOAD_ASYNC_TO_LDS_B128 (double-buffered, ASYNCcnt-ordered).
// ---------------------------------------------------------------------------

typedef __attribute__((ext_vector_type(16))) __bf16 v16bf;
typedef __attribute__((ext_vector_type(8)))  float  v8f;

#define DMODEL 1024
#define SEQ    2048
#define NH     16
#define DH     64
#define LOG2E  1.44269504f

struct U32x8 { uint4 lo, hi; };

__device__ __forceinline__ unsigned short f2bf(float f) {
  unsigned int u = __builtin_bit_cast(unsigned int, f);
  unsigned int r = u + 0x7fffu + ((u >> 16) & 1u);   // round-to-nearest-even
  return (unsigned short)(r >> 16);
}

// A/B^T fragment load for V_WMMA_*_16x16x32_BF16 (wave32 layout, ISA 7.12.2):
// lane l holds row m=l&15; half=l>>4 selects K in [8h,8h+8) and [16+8h,16+8h+8).
__device__ __forceinline__ v16bf load_frag(const unsigned short* __restrict__ base,
                                           int ld, int rowbase, int kbase, int lane) {
  const int m = lane & 15, half = lane >> 4;
  const unsigned short* p = base + (size_t)(rowbase + m) * ld + kbase + 8 * half;
  U32x8 t;
  t.lo = *(const uint4*)(p);
  t.hi = *(const uint4*)(p + 16);
  return __builtin_bit_cast(v16bf, t);
}

// Same fragment pattern from an LDS-resident tile with row stride `ld` (elems).
__device__ __forceinline__ v16bf load_frag_lds(const unsigned short* P, int ld,
                                               int rowbase, int lane) {
  const int m = lane & 15, half = lane >> 4;
  const unsigned short* p = P + (rowbase + m) * ld + 8 * half;
  U32x8 t;
  t.lo = *(const uint4*)(p);
  t.hi = *(const uint4*)(p + 16);
  return __builtin_bit_cast(v16bf, t);
}

__device__ __forceinline__ v8f wmma_bf16(v16bf a, v16bf b, v8f c) {
  return __builtin_amdgcn_wmma_f32_16x16x32_bf16(false, a, false, b,
                                                 (short)0, c, false, false);
}

// gfx1250 async copy: 16B/lane global -> LDS, tracked by ASYNCcnt.
__device__ __forceinline__ void async_g2l_b128(unsigned lds_off, const void* gaddr) {
  asm volatile("global_load_async_to_lds_b128 %0, %1, off"
               :: "v"(lds_off), "v"(gaddr) : "memory");
}
__device__ __forceinline__ void wait_asynccnt0() {
  asm volatile("s_wait_asynccnt 0x0" ::: "memory");
}

// ---------------------------------------------------------------------------
__global__ void cvt_f32_bf16(const float* __restrict__ src,
                             unsigned short* __restrict__ dst, int n) {
  int i = blockIdx.x * blockDim.x + threadIdx.x;
  if (i < n) dst[i] = f2bf(src[i]);
}

__global__ void bm_clear(unsigned int* __restrict__ bm, int n) {
  int i = blockIdx.x * blockDim.x + threadIdx.x;
  if (i < n) bm[i] = 0u;
}

__global__ void bm_scatter(const long long* __restrict__ inds, int n,
                           unsigned int* __restrict__ bm) {
  int i = blockIdx.x * blockDim.x + threadIdx.x;
  if (i >= n) return;
  int q = (int)inds[i];
  int k = (int)inds[n + i];
  atomicOr(&bm[q * (SEQ / 32) + (k >> 5)], 1u << (k & 31));
}

// ---------------------------------------------------------------------------
// C = A[M,1024] * W[1024,1024]^T, both K-contiguous (NT GEMM).
// Block: 8 waves, 256 rows x 64 cols. Wave: 32x64 (2 M-tiles x 4 N-tiles).
// W slab (64 rows x 32 K, 4KB) async-staged to LDS, double-buffered; each
// wave contributes one b128 async op (8 rows) per K-step.
// mode 0: bf16 row-major out.  mode 1: per-head transposed Vt[b,h,d,s].
// mode 2: fp32 out = X + acc (residual).
__global__ __launch_bounds__(256) void gemm_nt(const unsigned short* __restrict__ A,
                                               const unsigned short* __restrict__ W,
                                               unsigned short* __restrict__ Obf,
                                               const float* __restrict__ X,
                                               float* __restrict__ Of,
                                               int mode) {
  __shared__ __align__(16) unsigned short Wslab[2][64 * 32];

  const int lane = threadIdx.x & 31;
  const int wave = threadIdx.x >> 5;
  const int bn = blockIdx.x & 15;          // 16 N-blocks of 64
  const int bm_ = blockIdx.x >> 4;         // 32 M-blocks of 256
  const int nbase = bn * 64;
  const int mbase_w = bm_ * 256 + wave * 32;

  // this lane's share of the slab copy: row = wave*8 + lane>>2, 16B chunk lane&3
  const int srow = wave * 8 + (lane >> 2);
  const int schunk = (lane & 3) * 8;                    // element offset
  const unsigned short* gsrc0 = W + (size_t)(nbase + srow) * DMODEL + schunk;
  const unsigned lds0 = (unsigned)(size_t)(void*)&Wslab[0][0];
  const unsigned my_lds = lds0 + (unsigned)((srow * 32 + schunk) * 2);

  v8f acc[2][4] = {};

  // prologue: stage slab 0
  async_g2l_b128(my_lds, gsrc0);

  for (int kb = 0; kb < DMODEL; kb += 32) {
    const int buf = (kb >> 5) & 1;
    wait_asynccnt0();        // my rows of slab `buf` are in LDS
    __syncthreads();         // everyone's rows are in LDS; prev buf fully read
    if (kb + 32 < DMODEL)    // stage next slab into the other buffer
      async_g2l_b128(my_lds + (buf ^ 1) * (64 * 32 * 2), gsrc0 + kb + 32);

    v16bf a0 = load_frag(A, DMODEL, mbase_w, kb, lane);
    v16bf a1 = load_frag(A, DMODEL, mbase_w + 16, kb, lane);
    if (kb + 64 < DMODEL)
      __builtin_prefetch(A + (size_t)(mbase_w + (lane & 15)) * DMODEL + kb + 64, 0, 0);

    const unsigned short* sl = &Wslab[buf][0];
    for (int g = 0; g < 4; ++g) {
      v16bf bfrag = load_frag_lds(sl, 32, g * 16, lane);
      acc[0][g] = wmma_bf16(a0, bfrag, acc[0][g]);
      acc[1][g] = wmma_bf16(a1, bfrag, acc[1][g]);
    }
  }

  const int half = lane >> 4, n = lane & 15;
  for (int mi = 0; mi < 2; ++mi) {
    const int mb = mbase_w + mi * 16;
    for (int g = 0; g < 4; ++g) {
      const int nb = nbase + g * 16;
      v8f a = acc[mi][g];
      for (int i = 0; i < 8; ++i) {
        const int m = mb + i + 8 * half;
        if (mode == 0) {
          Obf[(size_t)m * DMODEL + nb + n] = f2bf(a[i]);
        } else if (mode == 1) {
          const int b_ = m >> 11, s = m & (SEQ - 1);
          const int col = nb + n;
          const int h = col >> 6, d = col & (DH - 1);
          Obf[(size_t)((b_ * NH + h) * DH + d) * SEQ + s] = f2bf(a[i]);
        } else {
          const size_t idx = (size_t)m * DMODEL + nb + n;
          Of[idx] = X[idx] + a[i];
        }
      }
    }
  }
}

// ---------------------------------------------------------------------------
// Flash attention: one wave per (b, h, 16-row q tile). K-dim 64 => 2 WMMAs per
// score tile; 32 k-columns per step; P staged via LDS for the PV WMMAs.
__global__ __launch_bounds__(32) void attn_flash(const unsigned short* __restrict__ Qb,
                                                 const unsigned short* __restrict__ Kb,
                                                 const unsigned short* __restrict__ Vt,
                                                 const unsigned int* __restrict__ bm,
                                                 unsigned short* __restrict__ Zb) {
  __shared__ __align__(16) unsigned short Pst[2][16 * 32];

  const int lane = threadIdx.x;
  const int qt = blockIdx.x & 127;
  const int h  = (blockIdx.x >> 7) & (NH - 1);
  const int b  = blockIdx.x >> 11;
  const int qbase = qt * 16;
  const int half = lane >> 4, n = lane & 15;

  const unsigned short* Qp = Qb + (size_t)(b * SEQ) * DMODEL + h * DH;
  const unsigned short* Kp = Kb + (size_t)(b * SEQ) * DMODEL + h * DH;
  const unsigned short* Vp = Vt + (size_t)((b * NH + h) * DH) * SEQ;

  const v16bf qa0 = load_frag(Qp, DMODEL, qbase, 0, lane);
  const v16bf qa1 = load_frag(Qp, DMODEL, qbase, 32, lane);

  float mrow[8], lrow[8];
  for (int i = 0; i < 8; ++i) { mrow[i] = -3.0e38f; lrow[i] = 0.0f; }
  v8f acc[4] = {};

  for (int kt = 0; kt < SEQ; kt += 32) {
    // ---- scores: two 16x16 tiles over 32 k-columns ----
    v16bf kb0 = load_frag(Kp, DMODEL, kt, 0, lane);
    v16bf kb1 = load_frag(Kp, DMODEL, kt, 32, lane);
    v16bf kb2 = load_frag(Kp, DMODEL, kt + 16, 0, lane);
    v16bf kb3 = load_frag(Kp, DMODEL, kt + 16, 32, lane);
    v8f s0 = {}, s1 = {};
    s0 = wmma_bf16(qa0, kb0, s0);
    s0 = wmma_bf16(qa1, kb1, s0);
    s1 = wmma_bf16(qa0, kb2, s1);
    s1 = wmma_bf16(qa1, kb3, s1);

    // ---- mask (pre-scale, as in reference), scale, online softmax ----
    const unsigned wcol = (unsigned)(kt >> 5);
    float p0[8], p1[8], csc[8];
    for (int i = 0; i < 8; ++i) {
      const int q = qbase + i + 8 * half;
      const unsigned w = bm[(size_t)q * (SEQ / 32) + wcol];
      float t0 = ((w >> n) & 1u)        ? -1.0e9f : s0[i];
      float t1 = ((w >> (n + 16)) & 1u) ? -1.0e9f : s1[i];
      t0 *= 0.125f;  // 1/sqrt(64)
      t1 *= 0.125f;
      float r = fmaxf(t0, t1);
      r = fmaxf(r, __shfl_xor(r, 1, 32));
      r = fmaxf(r, __shfl_xor(r, 2, 32));
      r = fmaxf(r, __shfl_xor(r, 4, 32));
      r = fmaxf(r, __shfl_xor(r, 8, 32));
      const float mnew = fmaxf(mrow[i], r);
      const float c  = exp2f((mrow[i] - mnew) * LOG2E);
      const float e0 = exp2f((t0 - mnew) * LOG2E);
      const float e1 = exp2f((t1 - mnew) * LOG2E);
      float rs = e0 + e1;
      rs += __shfl_xor(rs, 1, 32);
      rs += __shfl_xor(rs, 2, 32);
      rs += __shfl_xor(rs, 4, 32);
      rs += __shfl_xor(rs, 8, 32);
      lrow[i] = lrow[i] * c + rs;
      mrow[i] = mnew;
      csc[i] = c;
      p0[i] = e0;
      p1[i] = e1;
    }
    for (int g = 0; g < 4; ++g)
      for (int i = 0; i < 8; ++i) acc[g][i] *= csc[i];

    // ---- stage P (16x32 bf16) in LDS, re-read as A fragment ----
    unsigned short* P = &Pst[(kt >> 5) & 1][0];
    for (int i = 0; i < 8; ++i) {
      const int m = i + 8 * half;
      P[m * 32 + n]      = f2bf(p0[i]);
      P[m * 32 + 16 + n] = f2bf(p1[i]);
    }
    __syncthreads();
    const v16bf pf = load_frag_lds(P, 32, 0, lane);

    // ---- PV: z[16 x 64] accumulate, B-operand from per-head V^T rows ----
    for (int g = 0; g < 4; ++g) {
      v16bf vf = load_frag(Vp, SEQ, g * 16, kt, lane);
      acc[g] = wmma_bf16(pf, vf, acc[g]);
    }
  }

  // ---- normalize and emit z (bf16, row-major [B*S, 1024]) ----
  for (int i = 0; i < 8; ++i) {
    const float inv = 1.0f / lrow[i];
    const int m = qbase + i + 8 * half;
    const size_t rowoff = ((size_t)(b * SEQ + m)) * DMODEL + h * DH;
    for (int g = 0; g < 4; ++g)
      Zb[rowoff + g * 16 + n] = f2bf(acc[g][i] * inv);
  }
}

// ---------------------------------------------------------------------------
extern "C" void kernel_launch(void* const* d_in, const int* in_sizes, int n_in,
                              void* d_out, int out_size, void* d_ws, size_t ws_size,
                              hipStream_t stream) {
  const float*     x  = (const float*)d_in[0];
  const long long* mi = (const long long*)d_in[1];   // int64 [2, 100000]
  const float*     Wq = (const float*)d_in[2];
  const float*     Wk = (const float*)d_in[3];
  const float*     Wv = (const float*)d_in[4];
  const float*     Wo = (const float*)d_in[5];
  float* out = (float*)d_out;

  const int M  = in_sizes[0] / DMODEL;   // B*S = 8192
  const int NW = DMODEL * DMODEL;        // 1048576
  const int NX = in_sizes[0];            // 8388608
  const int NMASK = in_sizes[1] / 2;     // 100000
  const int NBMW  = SEQ * (SEQ / 32);    // 131072 bitmap words

  // workspace carve-out (256B aligned)
  char* ws = (char*)d_ws;
  size_t off = 0;
  auto carve = [&](size_t bytes) -> char* {
    char* p = ws + off;
    off = (off + bytes + 255) & ~(size_t)255;
    return p;
  };
  unsigned short* xb  = (unsigned short*)carve((size_t)NX * 2);
  unsigned short* wqb = (unsigned short*)carve((size_t)NW * 2);
  unsigned short* wkb = (unsigned short*)carve((size_t)NW * 2);
  unsigned short* wvb = (unsigned short*)carve((size_t)NW * 2);
  unsigned short* wob = (unsigned short*)carve((size_t)NW * 2);
  unsigned short* Qb  = (unsigned short*)carve((size_t)NX * 2);
  unsigned short* Kb  = (unsigned short*)carve((size_t)NX * 2);
  unsigned short* Vt  = (unsigned short*)carve((size_t)NX * 2);
  unsigned short* Zb  = (unsigned short*)carve((size_t)NX * 2);
  unsigned int*   bm  = (unsigned int*)carve((size_t)NBMW * 4);

  // 1) fp32 -> bf16 conversions
  cvt_f32_bf16<<<(NX + 255) / 256, 256, 0, stream>>>(x, xb, NX);
  cvt_f32_bf16<<<(NW + 255) / 256, 256, 0, stream>>>(Wq, wqb, NW);
  cvt_f32_bf16<<<(NW + 255) / 256, 256, 0, stream>>>(Wk, wkb, NW);
  cvt_f32_bf16<<<(NW + 255) / 256, 256, 0, stream>>>(Wv, wvb, NW);
  cvt_f32_bf16<<<(NW + 255) / 256, 256, 0, stream>>>(Wo, wob, NW);

  // 2) mask bitmap
  bm_clear<<<(NBMW + 255) / 256, 256, 0, stream>>>(bm, NBMW);
  bm_scatter<<<(NMASK + 255) / 256, 256, 0, stream>>>(mi, NMASK, bm);

  // 3) projections (V written transposed per head); 256x64 block tiles
  const int gemm_blocks = (M / 256) * (DMODEL / 64);   // 32*16 = 512
  gemm_nt<<<gemm_blocks, 256, 0, stream>>>(xb, wqb, Qb, nullptr, nullptr, 0);
  gemm_nt<<<gemm_blocks, 256, 0, stream>>>(xb, wkb, Kb, nullptr, nullptr, 0);
  gemm_nt<<<gemm_blocks, 256, 0, stream>>>(xb, wvb, Vt, nullptr, nullptr, 1);

  // 4) fused masked flash-attention
  const int attn_blocks = (M / SEQ) * NH * (SEQ / 16);    // 4*16*128 = 8192
  attn_flash<<<attn_blocks, 32, 0, stream>>>(Qb, Kb, Vt, bm, Zb);

  // 5) output projection + residual
  gemm_nt<<<gemm_blocks, 256, 0, stream>>>(Zb, wob, nullptr, x, out, 2);
}
